// memoryModel_28071906246972
// MI455X (gfx1250) — matrix operand order repaired and checked by reading, not compiled, not run on hardware
//
#include <hip/hip_runtime.h>
#include <hip/hip_bf16.h>

#define ERR_W 1.0f
#define EFF_W 0.1f
#define SPARS_W 0.05f
#define HIST_W 0.5f
#define RET_W 0.5f

#define WT_STRIDE 328   // padded row stride (bf16 elems): 164 dwords -> conflict-free DS reads
#define WT_BYTES (128 * WT_STRIDE * 2)   // 83968

typedef __attribute__((ext_vector_type(16))) __bf16 v16bf;
typedef __attribute__((ext_vector_type(8)))  __bf16 v8bf;
typedef __attribute__((ext_vector_type(8)))  float  v8f;

// ---------------------------------------------------------------------------
// Kernel 1: power iteration for ||D||_2^2, ||H||_2^2 -> step size; build the
// folded weight matrix W_t[n][k] (transposed, bf16, padded rows) in workspace.
//   k in [0,128)   : A   = I*(1 - step*(EFF+RET)) - step*(ERR*D^T D + HIST*H^T H)
//   k in [128,192) : B1  = step*ERR*D
//   k in [192,320) : B2  = step*(HIST*H + RET*I)
// ws_f[0] = threshold = step * SPARS_W
// ---------------------------------------------------------------------------
__global__ __launch_bounds__(128) void mm_setup_kernel(
    const float* __restrict__ D, const float* __restrict__ Hm,
    float* __restrict__ wsf, __bf16* __restrict__ wt)
{
    __shared__ float v[128];
    __shared__ float u[128];
    __shared__ float red[129];
    const int t = threadIdx.x;           // 0..127

    float lam[2];
    for (int which = 0; which < 2; ++which) {
        const float* M = which ? Hm : D;
        const int rows = which ? 128 : 64;     // M is [rows x 128]
        v[t] = 1.0f + 0.01f * (float)t;        // non-degenerate start vector
        __syncthreads();
        float lambda = 1.0f;
        for (int it = 0; it < 250; ++it) {
            if (t < rows) {                    // u = M v
                const float* mr = M + (size_t)t * 128;
                float s = 0.0f;
                for (int j = 0; j < 128; ++j) s = fmaf(mr[j], v[j], s);
                u[t] = s;
            }
            __syncthreads();
            float w = 0.0f;                    // w = M^T u  (Gram apply)
            for (int j = 0; j < rows; ++j) w = fmaf(M[(size_t)j * 128 + t], u[j], w);
            red[t] = w * w;
            __syncthreads();
            if (t == 0) {
                float s = 0.0f;
                for (int j = 0; j < 128; ++j) s += red[j];
                red[128] = sqrtf(s);
            }
            __syncthreads();
            lambda = red[128];                 // ||G v|| -> lambda_max
            v[t] = w / lambda;
            __syncthreads();
        }
        lam[which] = lambda;
    }

    const float L    = ERR_W * lam[0] + EFF_W + HIST_W * lam[1] + RET_W;
    const float step = 1.0f / L;
    if (t == 0) wsf[0] = step * SPARS_W;

    const int n = t;                           // build W_t row n
    __bf16* row = wt + (size_t)n * WT_STRIDE;
    for (int m = 0; m < 128; ++m) {
        float gd = 0.0f;
        for (int j = 0; j < 64; ++j)
            gd = fmaf(D[(size_t)j * 128 + m], D[(size_t)j * 128 + n], gd);
        float gh = 0.0f;
        for (int j = 0; j < 128; ++j)
            gh = fmaf(Hm[(size_t)j * 128 + m], Hm[(size_t)j * 128 + n], gh);
        float val = -step * (ERR_W * gd + HIST_W * gh);
        if (m == n) val += 1.0f - step * (EFF_W + RET_W);
        row[m] = (__bf16)val;
    }
    for (int j = 0; j < 64; ++j)
        row[128 + j] = (__bf16)(step * ERR_W * D[(size_t)j * 128 + n]);
    for (int m = 0; m < 128; ++m) {
        float val = step * HIST_W * Hm[(size_t)m * 128 + n];
        if (m == n) val += step * RET_W;
        row[192 + m] = (__bf16)val;
    }
    for (int k = 320; k < WT_STRIDE; ++k) row[k] = (__bf16)0.0f;  // pad
}

// ---------------------------------------------------------------------------
// Kernel 2: out[B,128] = softthresh( [state | stim | ref] @ W , thr )
// 256 threads = 8 waves; each wave: 16 batch rows x full 128 output columns.
// W_t is async-copied to LDS once per workgroup (ASYNCcnt path); B fragments
// then come from LDS via ds_load_b128 with a conflict-free padded stride, so
// the VMEM pipe carries only the streaming fp32 A reads + the output stores.
// ---------------------------------------------------------------------------
__global__ __launch_bounds__(256) void mm_pgm_kernel(
    const float* __restrict__ stim,   // [B,64]
    const float* __restrict__ ref,    // [B,128]
    const float* __restrict__ state,  // [B,128]
    const float* __restrict__ wsf,    // [0] = threshold
    const __bf16* __restrict__ wt,    // W_t[128][WT_STRIDE] in global (from ws)
    float* __restrict__ out)          // [B,128]
{
    extern __shared__ __bf16 lds_w[]; // WT_BYTES

    const int tid  = threadIdx.x;
    const int lane = tid & 31;
    const int wave = tid >> 5;
    const int r    = lane & 15;       // A: row-in-tile ; B/D: col-in-tile
    const int hi   = lane >> 4;       // half-wave selector
    const int rowBase = blockIdx.x * 128 + wave * 16;

    // ---- Stage W_t into LDS with async global->LDS b128 copies ------------
    {
        const char* gsrc = (const char*)wt;
        char*       ldst = (char*)lds_w;
#pragma unroll
        for (int i = 0; i < 21; ++i) {
            unsigned off = (unsigned)tid * 16u + (unsigned)i * 4096u;
            if (off < (unsigned)WT_BYTES) {
                unsigned lds_addr = (unsigned)(unsigned long long)(ldst + off);
                unsigned long long gaddr = (unsigned long long)(gsrc + off);
                asm volatile("global_load_async_to_lds_b128 %0, %1, off"
                             :: "v"(lds_addr), "v"(gaddr) : "memory");
            }
        }
        asm volatile("s_wait_asynccnt 0x0" ::: "memory");
        __syncthreads();
    }

    v8f acc[8];
#pragma unroll
    for (int i = 0; i < 8; ++i) acc[i] = {};

#pragma unroll
    for (int kk = 0; kk < 10; ++kk) {
        // Select source slice of the virtual [x | s | r] concatenation
        const float* src; int stride, col;
        if (kk < 4)      { src = state; stride = 128; col = kk * 32; }
        else if (kk < 6) { src = stim;  stride = 64;  col = (kk - 4) * 32; }
        else             { src = ref;   stride = 128; col = (kk - 6) * 32; }

        // A fragment, 16-bit A 16x32 layout: lane holds row (lane&15);
        // low half-wave: K = col+{0..7} and col+16+{0..7}; high half: +8.
        const float* rp = src + (size_t)(rowBase + r) * stride + col + hi * 8;
        float4 f0 = *(const float4*)(rp);
        float4 f1 = *(const float4*)(rp + 4);
        float4 f2 = *(const float4*)(rp + 16);
        float4 f3 = *(const float4*)(rp + 20);
        v16bf a;
        a[0]  = (__bf16)f0.x; a[1]  = (__bf16)f0.y; a[2]  = (__bf16)f0.z; a[3]  = (__bf16)f0.w;
        a[4]  = (__bf16)f1.x; a[5]  = (__bf16)f1.y; a[6]  = (__bf16)f1.z; a[7]  = (__bf16)f1.w;
        a[8]  = (__bf16)f2.x; a[9]  = (__bf16)f2.y; a[10] = (__bf16)f2.z; a[11] = (__bf16)f2.w;
        a[12] = (__bf16)f3.x; a[13] = (__bf16)f3.y; a[14] = (__bf16)f3.z; a[15] = (__bf16)f3.w;

        const int kc = kk * 32;
#pragma unroll
        for (int ct = 0; ct < 8; ++ct) {
            // B fragment from LDS: 16-bit B 32x16 layout: lane holds column
            // (lane&15), K = kc+{0..7} and kc+16+{0..7}; high half: +8. Padded stride (164
            // dwords) puts the 16 lanes on 16 distinct 4-bank groups.
            const __bf16* wp = lds_w + (ct * 16 + r) * WT_STRIDE + kc + hi * 8;
            v8bf blo = *(const v8bf*)(wp);
            v8bf bhi = *(const v8bf*)(wp + 16);
            v16bf b = __builtin_shufflevector(blo, bhi,
                        0,1,2,3,4,5,6,7,8,9,10,11,12,13,14,15);
            acc[ct] = __builtin_amdgcn_wmma_f32_16x16x32_bf16(
                false, a, false, b, (short)0, acc[ct], false, false);
        }
    }

    const float thr = wsf[0];
#pragma unroll
    for (int ct = 0; ct < 8; ++ct) {
#pragma unroll
        for (int j = 0; j < 8; ++j) {
            // C/D f32 16x16 layout: VGPR j -> M = j + hi*8, N = lane&15
            float p = acc[ct][j];
            float y = fmaxf(p - thr, 0.0f) - fmaxf(-p - thr, 0.0f);
            out[(size_t)(rowBase + hi * 8 + j) * 128 + ct * 16 + r] = y;
        }
    }
}

// ---------------------------------------------------------------------------
extern "C" void kernel_launch(void* const* d_in, const int* in_sizes, int n_in,
                              void* d_out, int out_size, void* d_ws, size_t ws_size,
                              hipStream_t stream)
{
    const float* stim  = (const float*)d_in[0];  // [B,64]
    const float* ref   = (const float*)d_in[1];  // [B,128]
    const float* state = (const float*)d_in[2];  // [B,128]
    const float* D     = (const float*)d_in[3];  // [64,128]
    const float* Hm    = (const float*)d_in[4];  // [128,128]

    float*  wsf = (float*)d_ws;
    __bf16* wt  = (__bf16*)((char*)d_ws + 256);  // 128*328 bf16 ~ 82KB

    const int B    = in_sizes[2] / 128;          // rows (262144)
    const int grid = B / 128;                    // 128 rows per workgroup

    (void)hipFuncSetAttribute((const void*)mm_pgm_kernel,
                              hipFuncAttributeMaxDynamicSharedMemorySize,
                              WT_BYTES);

    mm_setup_kernel<<<1, 128, 0, stream>>>(D, Hm, wsf, wt);
    mm_pgm_kernel<<<grid, 256, WT_BYTES, stream>>>(stim, ref, state, wsf, wt,
                                                   (float*)d_out);
}
